// GraphGINLinkPredictor_31825707663445
// MI455X (gfx1250) — compile-verified, matrix operand order checked by reading, not run on hardware
//
#include <hip/hip_runtime.h>

typedef __attribute__((ext_vector_type(16))) _Float16 v16h;
typedef __attribute__((ext_vector_type(8)))  _Float16 v8h;
typedef __attribute__((ext_vector_type(8)))  float    v8f;

#define NNODES 50000
#define NEDGES 500000
#define NLBL   100000
#define HID    256
#define LN_EPS 1e-5f

static inline int ceil_div(int a, int b) { return (a + b - 1) / b; }

// ---------------------------------------------------------------- utilities

__global__ void k_zero_f32(float* __restrict__ p, int n) {
  int i = blockIdx.x * blockDim.x + threadIdx.x;
  if (i < n) p[i] = 0.f;
}

// WT[n*K + k] = (f16) W[k*N + n]
__global__ void k_transpose_f16(const float* __restrict__ W, _Float16* __restrict__ WT,
                                int K, int N) {
  int t = blockIdx.x * blockDim.x + threadIdx.x;
  if (t >= K * N) return;
  int k = t / N, n = t - k * N;
  WT[n * K + k] = (_Float16)W[t];
}

// ---------------------------------------------------------------- scatter-add

__global__ void k_scatter_add_d2(const float* __restrict__ h,
                                 const int* __restrict__ src,
                                 const int* __restrict__ dst,
                                 float* __restrict__ agg) {
  int e = blockIdx.x * blockDim.x + threadIdx.x;
  if (e >= NEDGES) return;
  int s = src[e], d = dst[e];
  float v0 = h[(size_t)s * 2 + 0];
  float v1 = h[(size_t)s * 2 + 1];
  __hip_atomic_fetch_add(agg + (size_t)d * 2 + 0, v0, __ATOMIC_RELAXED, __HIP_MEMORY_SCOPE_AGENT);
  __hip_atomic_fetch_add(agg + (size_t)d * 2 + 1, v1, __ATOMIC_RELAXED, __HIP_MEMORY_SCOPE_AGENT);
}

// 64 threads per edge, 4 dims each (float4 gather, 4 hw f32 atomics)
__global__ void k_scatter_add_d256(const float* __restrict__ h,
                                   const int* __restrict__ src,
                                   const int* __restrict__ dst,
                                   float* __restrict__ agg) {
  int t = blockIdx.x * blockDim.x + threadIdx.x;
  if (t >= NEDGES * 64) return;
  int e = t >> 6;
  int c = (t & 63) * 4;
  int s = src[e], d = dst[e];
  const float4 v = *(const float4*)(h + (size_t)s * HID + c);
  float* ap = agg + (size_t)d * HID + c;
  __hip_atomic_fetch_add(ap + 0, v.x, __ATOMIC_RELAXED, __HIP_MEMORY_SCOPE_AGENT);
  __hip_atomic_fetch_add(ap + 1, v.y, __ATOMIC_RELAXED, __HIP_MEMORY_SCOPE_AGENT);
  __hip_atomic_fetch_add(ap + 2, v.z, __ATOMIC_RELAXED, __HIP_MEMORY_SCOPE_AGENT);
  __hip_atomic_fetch_add(ap + 3, v.w, __ATOMIC_RELAXED, __HIP_MEMORY_SCOPE_AGENT);
}

// x16 = f16( (1+eps[l])*h + agg )
__global__ void k_combine_f16(const float* __restrict__ h, const float* __restrict__ agg,
                              const float* __restrict__ eps, int layer,
                              _Float16* __restrict__ out, int n) {
  int i = blockIdx.x * blockDim.x + threadIdx.x;
  if (i >= n) return;
  float ep = 1.0f + eps[layer];
  out[i] = (_Float16)(ep * h[i] + agg[i]);
}

// ---------------------------------------------------------------- layer-0 first MLP (K=2, VALU)

__global__ void k_layer0_mlp1(const float* __restrict__ x, const float* __restrict__ agg2,
                              const float* __restrict__ eps,
                              const float* __restrict__ W1, const float* __restrict__ b1,
                              _Float16* __restrict__ out) {
  int t = blockIdx.x * blockDim.x + threadIdx.x;
  if (t >= NNODES * HID) return;
  int n = t >> 8, j = t & 255;
  float ep = 1.0f + eps[0];
  float z0 = ep * x[(size_t)n * 2 + 0] + agg2[(size_t)n * 2 + 0];
  float z1 = ep * x[(size_t)n * 2 + 1] + agg2[(size_t)n * 2 + 1];
  float v = z0 * W1[j] + z1 * W1[HID + j] + b1[j];
  out[t] = (_Float16)fmaxf(v, 0.f);
}

// ---------------------------------------------------------------- WMMA GEMM
// C[M,Nout] = A[M,K] * BT[Nout,K]^T + bias ; optional ReLU ; f32 or f16 output.
// Block = 8 waves = TMB M-supertiles x NG N-groups (NG*64 output cols).
// A supertiles staged in LDS via global_load_async_to_lds_b128 (ASYNCcnt),
// fragments then read with ds_load_b128. B read from global (L2-resident).
// 16-bit fragment layout (ISA 7.12.2): lanes 0-15 hold K=kb..kb+7 (halfs 0-7)
// and K=kb+16..kb+23 (halfs 8-15) with kb=k0; lanes 16-31 use kb=k0+8.

__device__ __forceinline__ v16h load_frag(const _Float16* __restrict__ row, int kb) {
  v8h lo = *(const v8h*)(row + kb);
  v8h hi = *(const v8h*)(row + kb + 16);
  v16h f;
#pragma unroll
  for (int i = 0; i < 8; ++i) { f[i] = lo[i]; f[8 + i] = hi[i]; }
  return f;
}

template <int K, int NG, bool RELU, bool OUT16>
__global__ void k_gemm_wmma(const _Float16* __restrict__ A, const _Float16* __restrict__ BT,
                            const float* __restrict__ bias, void* __restrict__ Cout,
                            int Mtiles) {
  constexpr int TMB = 8 / NG;                       // M supertiles per block
  __shared__ __align__(16) _Float16 shA[TMB * 16 * K];

  int wave = threadIdx.x >> 5;
  int lane = threadIdx.x & 31;
  int tm0 = blockIdx.x * TMB;

  // ---- cooperative async copy: TMB contiguous 16-row A strips -> LDS ----
  {
    const char* gbase = (const char*)(A + (size_t)tm0 * 16 * K);
    long long remain = ((long long)Mtiles - tm0) * 16 * K * 2 - 16;  // last valid 16B chunk
    unsigned ldsbase = (unsigned)(uintptr_t)&shA[0];
    constexpr int CHUNKS = (TMB * 16 * K * 2) / (256 * 16);
#pragma unroll
    for (int it = 0; it < CHUNKS; ++it) {
      long long off = (long long)(it * 256 + threadIdx.x) * 16;
      long long goff = off > remain ? remain : off;  // clamp in-bounds (tail block)
      unsigned daddr = ldsbase + (unsigned)off;
      unsigned long long gaddr = (unsigned long long)(gbase + goff);
      asm volatile("global_load_async_to_lds_b128 %0, %1, off"
                   :: "v"(daddr), "v"(gaddr) : "memory");
    }
    asm volatile("s_wait_asynccnt 0x0" ::: "memory");
  }
  __syncthreads();

  int tmL = wave / NG;
  int tg  = wave % NG;
  int tm  = tm0 + tmL;
  if (tm >= Mtiles) return;                     // wave-uniform, after barrier

  int lrow = lane & 15;
  int lhi  = lane >> 4;                         // 0 or 1

  const _Float16* Ar = shA + (size_t)(tmL * 16 + lrow) * K;     // LDS
  const _Float16* Br = BT  + (size_t)(tg * 64 + lrow) * K;      // global

  v8f acc0 = {}, acc1 = {}, acc2 = {}, acc3 = {};
#pragma unroll
  for (int k0 = 0; k0 < K; k0 += 32) {
    int kb = k0 + lhi * 8;
    v16h af = load_frag(Ar, kb);
    v16h b0 = load_frag(Br + (size_t)0 * 16 * K, kb);
    v16h b1 = load_frag(Br + (size_t)1 * 16 * K, kb);
    v16h b2 = load_frag(Br + (size_t)2 * 16 * K, kb);
    v16h b3 = load_frag(Br + (size_t)3 * 16 * K, kb);
    acc0 = __builtin_amdgcn_wmma_f32_16x16x32_f16(false, af, false, b0, (short)0, acc0, false, false);
    acc1 = __builtin_amdgcn_wmma_f32_16x16x32_f16(false, af, false, b1, (short)0, acc1, false, false);
    acc2 = __builtin_amdgcn_wmma_f32_16x16x32_f16(false, af, false, b2, (short)0, acc2, false, false);
    acc3 = __builtin_amdgcn_wmma_f32_16x16x32_f16(false, af, false, b3, (short)0, acc3, false, false);
  }

  // C/D layout: VGPR i -> row (i + 8*lhi), col = lrow (within tile)
  constexpr int Nout = NG * 64;
  v8f accs[4] = {acc0, acc1, acc2, acc3};
#pragma unroll
  for (int j = 0; j < 4; ++j) {
    int col = (tg * 4 + j) * 16 + lrow;
    float bv = bias[col];
#pragma unroll
    for (int i = 0; i < 8; ++i) {
      float v = accs[j][i] + bv;
      if (RELU) v = fmaxf(v, 0.f);
      size_t idxo = (size_t)(tm * 16 + i + lhi * 8) * Nout + col;
      if (OUT16) ((_Float16*)Cout)[idxo] = (_Float16)v;
      else       ((float*)Cout)[idxo] = v;
    }
  }
}

// ---------------------------------------------------------------- layernorm (wave32 per row)

__global__ void k_layernorm(const float* __restrict__ z, const float* __restrict__ gamma,
                            const float* __restrict__ beta, float* __restrict__ out, int rows) {
  int wave = threadIdx.x >> 5, lane = threadIdx.x & 31;
  int row = blockIdx.x * (blockDim.x >> 5) + wave;
  if (row >= rows) return;
  const float* zr = z + (size_t)row * HID;
  float v[8];
  float s = 0.f;
#pragma unroll
  for (int i = 0; i < 8; ++i) { v[i] = zr[lane + i * 32]; s += v[i]; }
#pragma unroll
  for (int m = 16; m >= 1; m >>= 1) s += __shfl_xor(s, m, 32);
  float mu = s * (1.0f / HID);
  float q = 0.f;
#pragma unroll
  for (int i = 0; i < 8; ++i) { float d = v[i] - mu; q += d * d; }
#pragma unroll
  for (int m = 16; m >= 1; m >>= 1) q += __shfl_xor(q, m, 32);
  float inv = rsqrtf(q * (1.0f / HID) + LN_EPS);
  float* orow = out + (size_t)row * HID;
#pragma unroll
  for (int i = 0; i < 8; ++i) {
    int c = lane + i * 32;
    orow[c] = gamma[c] * (v[i] - mu) * inv + beta[c];
  }
}

// ---------------------------------------------------------------- edge feature gather

__global__ void k_edge_gather(const float* __restrict__ h, const int* __restrict__ sidx,
                              const int* __restrict__ didx, _Float16* __restrict__ ef) {
  int t = blockIdx.x * blockDim.x + threadIdx.x;
  if (t >= NLBL * 512) return;
  int e = t >> 9;
  int j = t & 511;
  int n = (j < HID) ? sidx[e] : didx[e];
  int c = j & 255;
  ef[t] = (_Float16)h[(size_t)n * HID + c];
}

// ---------------------------------------------------------------- final dot (128 -> 1)

__global__ void k_final_dot(const _Float16* __restrict__ e2, const float* __restrict__ W3,
                            const float* __restrict__ b3, float* __restrict__ out) {
  int e = blockIdx.x * blockDim.x + threadIdx.x;
  if (e >= NLBL) return;
  const _Float16* r = e2 + (size_t)e * 128;
  float s = 0.f;
#pragma unroll
  for (int j = 0; j < 128; ++j) s += (float)r[j] * W3[j];
  out[e] = s + b3[0];
}

// ---------------------------------------------------------------- host launcher

extern "C" void kernel_launch(void* const* d_in, const int* in_sizes, int n_in,
                              void* d_out, int out_size, void* d_ws, size_t ws_size,
                              hipStream_t stream) {
  (void)in_sizes; (void)n_in; (void)out_size; (void)ws_size;

  const float* x   = (const float*)d_in[0];
  const int*   ei  = (const int*)d_in[1];
  const int*   src = ei;
  const int*   dst = ei + NEDGES;
  const int*   eli = (const int*)d_in[2];
  const int*   ls  = eli;
  const int*   ld  = eli + NLBL;

  const float *W1[3], *b1[3], *W2[3], *b2[3], *gam[3], *bet[3];
  int idx = 3;
  for (int l = 0; l < 3; ++l) {
    W1[l] = (const float*)d_in[idx++]; b1[l] = (const float*)d_in[idx++];
    W2[l] = (const float*)d_in[idx++]; b2[l] = (const float*)d_in[idx++];
  }
  for (int l = 0; l < 3; ++l) {
    gam[l] = (const float*)d_in[idx++]; bet[l] = (const float*)d_in[idx++];
  }
  const float* eps = (const float*)d_in[idx++];
  const float* We1 = (const float*)d_in[idx++];
  const float* be1 = (const float*)d_in[idx++];
  const float* We2 = (const float*)d_in[idx++];
  const float* be2 = (const float*)d_in[idx++];
  const float* We3 = (const float*)d_in[idx++];
  const float* be3 = (const float*)d_in[idx++];

  // workspace layout
  const size_t P  = (size_t)NNODES * HID * sizeof(float);   // 51.2 MB f32 plane
  const size_t Ph = (size_t)NNODES * HID * sizeof(_Float16);// 25.6 MB f16 plane
  char* ws = (char*)d_ws;
  float*    h    = (float*)(ws);              // P0
  float*    agg  = (float*)(ws + P);          // P1
  float*    z    = (float*)(ws + 2 * P);      // P2
  _Float16* x16a = (_Float16*)(ws + 3 * P);           // P3
  _Float16* x16b = (_Float16*)(ws + 3 * P + Ph);      // P4
  _Float16* wtA  = (_Float16*)(ws + 3 * P + 2 * Ph);  // up to 256x512 f16
  _Float16* wtB  = wtA + 256 * 512;                   // up to 256x256 f16
  // edge-phase overlays (GNN buffers dead by then)
  _Float16* ef16 = (_Float16*)(ws + P);   // [NLBL,512] spans P1..P2
  _Float16* e1   = (_Float16*)(ws);       // [NLBL,256] in P0
  _Float16* e2   = x16a;                  // [NLBL,128] in P3
  float*    logits = (float*)d_out;

  const int T = 256;
  const int WAVES = T / 32;

  // ---------------- layer 0 ----------------
  k_zero_f32<<<ceil_div(NNODES * 2, T), T, 0, stream>>>(agg, NNODES * 2);
  k_scatter_add_d2<<<ceil_div(NEDGES, T), T, 0, stream>>>(x, src, dst, agg);
  k_layer0_mlp1<<<ceil_div(NNODES * HID, T), T, 0, stream>>>(x, agg, eps, W1[0], b1[0], x16a);
  k_transpose_f16<<<ceil_div(HID * HID, T), T, 0, stream>>>(W2[0], wtB, HID, HID);
  k_gemm_wmma<HID, 4, false, false><<<ceil_div(NNODES / 16, 2), T, 0, stream>>>(
      x16a, wtB, b2[0], z, NNODES / 16);
  k_layernorm<<<ceil_div(NNODES, WAVES), T, 0, stream>>>(z, gam[0], bet[0], h, NNODES);

  // ---------------- layers 1,2 ----------------
  for (int l = 1; l < 3; ++l) {
    k_zero_f32<<<ceil_div(NNODES * HID, T), T, 0, stream>>>(agg, NNODES * HID);
    k_scatter_add_d256<<<ceil_div(NEDGES * 64, T), T, 0, stream>>>(h, src, dst, agg);
    k_combine_f16<<<ceil_div(NNODES * HID, T), T, 0, stream>>>(h, agg, eps, l, x16a, NNODES * HID);
    k_transpose_f16<<<ceil_div(HID * HID, T), T, 0, stream>>>(W1[l], wtA, HID, HID);
    k_transpose_f16<<<ceil_div(HID * HID, T), T, 0, stream>>>(W2[l], wtB, HID, HID);
    k_gemm_wmma<HID, 4, true, true><<<ceil_div(NNODES / 16, 2), T, 0, stream>>>(
        x16a, wtA, b1[l], x16b, NNODES / 16);
    k_gemm_wmma<HID, 4, false, false><<<ceil_div(NNODES / 16, 2), T, 0, stream>>>(
        x16b, wtB, b2[l], z, NNODES / 16);
    k_layernorm<<<ceil_div(NNODES, WAVES), T, 0, stream>>>(z, gam[l], bet[l], h, NNODES);
  }

  // ---------------- edge MLP ----------------
  k_edge_gather<<<ceil_div(NLBL * 512, T), T, 0, stream>>>(h, ls, ld, ef16);
  k_transpose_f16<<<ceil_div(512 * HID, T), T, 0, stream>>>(We1, wtA, 512, HID); // WT[256,512]
  k_gemm_wmma<512, 4, true, true><<<ceil_div(NLBL / 16, 2), T, 0, stream>>>(
      ef16, wtA, be1, e1, NLBL / 16);
  k_transpose_f16<<<ceil_div(HID * 128, T), T, 0, stream>>>(We2, wtB, HID, 128); // WT[128,256]
  k_gemm_wmma<HID, 2, true, true><<<ceil_div(NLBL / 16, 4), T, 0, stream>>>(
      e1, wtB, be2, e2, NLBL / 16);
  k_final_dot<<<ceil_div(NLBL, T), T, 0, stream>>>(e2, We3, be3, logits);
}
